// MultiHeadAttention_2224793060021
// MI455X (gfx1250) — matrix operand order repaired and checked by reading, not compiled, not run on hardware
//
#include <hip/hip_runtime.h>

#define DEV_INLINE __device__ __forceinline__

typedef __attribute__((ext_vector_type(16))) __bf16 v16bf;
typedef __attribute__((ext_vector_type(8)))  __bf16 v8bf;
typedef __attribute__((ext_vector_type(8)))  float  v8f;

constexpr int B = 4, S = 2048, D = 512, H = 8, DK = 64, DV = 64;
constexpr int HD = H * DK;       // 512
constexpr int SPAD = S + 4;      // LDS row stride 2052 dwords: 16B-aligned, conflict-free

DEV_INLINE v8f wmma_bf16(v16bf a, v16bf b, v8f c) {
  return __builtin_amdgcn_wmma_f32_16x16x32_bf16(false, a, false, b, (short)0, c, false, false);
}

DEV_INLINE v16bf cvt16(const float4& a, const float4& b, const float4& c, const float4& d) {
  v16bf r;
  r[0]=(__bf16)a.x;  r[1]=(__bf16)a.y;  r[2]=(__bf16)a.z;  r[3]=(__bf16)a.w;
  r[4]=(__bf16)b.x;  r[5]=(__bf16)b.y;  r[6]=(__bf16)b.z;  r[7]=(__bf16)b.w;
  r[8]=(__bf16)c.x;  r[9]=(__bf16)c.y;  r[10]=(__bf16)c.z; r[11]=(__bf16)c.w;
  r[12]=(__bf16)d.x; r[13]=(__bf16)d.y; r[14]=(__bf16)d.z; r[15]=(__bf16)d.w;
  return r;
}

// LDS byte-offset of a __shared__ object (for async-to-LDS VDST operand)
DEV_INLINE unsigned lds_offset(const void* p) {
  return (unsigned)(size_t)(__attribute__((address_space(3))) const void*)p;
}

// ---------------------------------------------------------------------------
// W transpose: W [D, HD] f32 -> Wt [HD, D] bf16.
// Staging uses the CDNA5 async global->LDS DMA path (ASYNCcnt).
// ---------------------------------------------------------------------------
__global__ __launch_bounds__(256, 4)
void transpose_w(const float* __restrict__ W, __bf16* __restrict__ Wt) {
  __shared__ float t[64][68];                 // row stride 272B: 16B-aligned chunks
  const int k0 = blockIdx.x * 64, n0 = blockIdx.y * 64;
#pragma unroll
  for (int p = 0; p < 4; ++p) {
    int i = p * 256 + threadIdx.x;            // 1024 x 16B chunks = 16KB tile
    int r = i >> 4, c = (i & 15) << 2;
    const float* gsrc = W + (size_t)(k0 + r) * HD + n0 + c;   // coalesced over c
    unsigned loff = lds_offset(&t[r][c]);
    asm volatile("global_load_async_to_lds_b128 %0, %1, off"
                 :: "v"(loff), "v"(gsrc) : "memory");
  }
  asm volatile("s_wait_asynccnt 0x0" ::: "memory");
  __syncthreads();
  for (int i = threadIdx.x; i < 64 * 64; i += 256) {
    int r = i >> 6, c = i & 63;
    Wt[(size_t)(n0 + r) * D + k0 + c] = (__bf16)t[c][r];      // coalesced over c
  }
}

// ---------------------------------------------------------------------------
// Projection GEMM, LDS-free: out tile 16x64 per 128-thread block (wave = 16x16).
// vmode=0: store [b,h,s,64] bf16 (Q,K).  vmode=1: store transposed [b,h,d,S] (V).
// ---------------------------------------------------------------------------
__global__ __launch_bounds__(128, 4)
void proj_kernel(const float* __restrict__ X, const __bf16* __restrict__ Wt,
                 __bf16* __restrict__ Out, float scale, int vmode) {
  const int row0 = blockIdx.x * 16;
  const int cg   = blockIdx.y;
  const int wave = threadIdx.x >> 5;
  const int lane = threadIdx.x & 31;
  const int m    = lane & 15;
  const int g    = lane >> 4;

  const float*  xrow = X  + (size_t)(row0 + m) * D;
  const __bf16* wrow = Wt + (size_t)(cg * 64 + wave * 16 + m) * D;

  v8f acc = {};
#pragma unroll 2
  for (int k0 = 0; k0 < D; k0 += 32) {
    float4 x0 = *(const float4*)(xrow + k0 + 16 * g);
    float4 x1 = *(const float4*)(xrow + k0 + 16 * g + 4);
    float4 x2 = *(const float4*)(xrow + k0 + 16 * g + 8);
    float4 x3 = *(const float4*)(xrow + k0 + 16 * g + 12);
    v16bf a = cvt16(x0, x1, x2, x3);
    v16bf b = *(const v16bf*)(wrow + k0 + 16 * g);   // 16 contiguous K elems
    acc = wmma_bf16(a, b, acc);
  }

  const int mb  = g * 8;
  const int col = cg * 64 + wave * 16 + m;
  const int h   = col >> 6, d = col & 63;
  if (vmode == 0) {
#pragma unroll
    for (int i = 0; i < 8; ++i) {
      int row = row0 + mb + i;
      int bi = row >> 11, s = row & (S - 1);
      Out[(((size_t)(bi * H + h) * S + s) << 6) + d] = (__bf16)(acc[i] * scale);
    }
  } else {
    int row = row0 + mb;
    int bi = row >> 11, s = row & (S - 1);
    v8bf vv;
#pragma unroll
    for (int i = 0; i < 8; ++i) vv[i] = (__bf16)(acc[i] * scale);
    *(v8bf*)(Out + (((size_t)(bi * H + h) << 6) + d) * S + s) = vv;   // one 16B store
  }
}

// ---------------------------------------------------------------------------
// Fused attention: block = (b, h, 16-row q tile), 256 threads = 8 waves.
//  1) raw scores via WMMA (mask prefetched, not loaded here)
//  2) softmax pass1 fuses + gamma*mask (coalesced float4 stream) with row max
//  3) exp+sum (float4); 4) attn writeback (float4); 5) attn @ V via WMMA
// ---------------------------------------------------------------------------
__global__ __launch_bounds__(256, 1)
void attn_kernel(const __bf16* __restrict__ Qh, const __bf16* __restrict__ Kh,
                 const __bf16* __restrict__ Vt, const float* __restrict__ mask,
                 const float* __restrict__ gamma_p,
                 float* __restrict__ out, float* __restrict__ attn_out) {
  __shared__ float sc[16][SPAD];
  __shared__ float outbuf[16][64];
  __shared__ float rowinv[16];

  const int q0 = blockIdx.x * 16;
  const int h  = blockIdx.y;
  const int bb = blockIdx.z;
  const int wave = threadIdx.x >> 5;
  const int lane = threadIdx.x & 31;
  const int m = lane & 15;
  const int g = lane >> 4;
  const int mb = g * 8;
  const float gamma = gamma_p[0];

  const size_t bh = (size_t)(bb * H + h);
  const __bf16* Qb = Qh + ((bh * S + q0) << 6);
  const __bf16* Kb = Kh + (bh * S << 6);
  const float* mstrip = mask + ((size_t)bb * S + q0) * S;   // 16 x S strip

  // Q fragments, loaded once (4 contiguous 16B chunks)
  v16bf aq0, aq1;
  {
    const __bf16* qrow = Qb + m * 64;
    v8bf qa = *(const v8bf*)(qrow + 16 * g);
    v8bf qb = *(const v8bf*)(qrow + 8 + 16 * g);
    v8bf qc = *(const v8bf*)(qrow + 32 + 16 * g);
    v8bf qd = *(const v8bf*)(qrow + 40 + 16 * g);
#pragma unroll
    for (int i = 0; i < 8; ++i) {
      aq0[i] = qa[i]; aq0[8 + i] = qb[i];
      aq1[i] = qc[i]; aq1[8 + i] = qd[i];
    }
  }

  // ---- 1) raw scores; prefetch this tile's mask block to warm L2 ----
  for (int t = wave; t < S / 16; t += 8) {
    const int k0 = t * 16;
    // 32 lanes cover the 16x16 mask block (rows m, 64B halves by g)
    __builtin_prefetch((const char*)(mstrip + (size_t)m * S + k0) + g * 64, 0, 1);
    const __bf16* krow = Kb + (size_t)(k0 + m) * 64;
    v16bf bk0 = *(const v16bf*)(krow + 16 * g);
    v16bf bk1 = *(const v16bf*)(krow + 32 + 16 * g);
    v8f c = {};
    c = wmma_bf16(aq0, bk0, c);
    c = wmma_bf16(aq1, bk1, c);
#pragma unroll
    for (int i = 0; i < 8; ++i) sc[mb + i][k0 + m] = c[i];
  }
  __syncthreads();

  // ---- 2+3) softmax: wave per row; pass1 fuses gamma*mask (float4 stream) ----
  for (int r = wave; r < 16; r += 8) {
    const float* mrow = mstrip + (size_t)r * S;
    float mx = -3.4e38f;
    for (int c = lane * 4; c < S; c += 128) {
      float4 sv = *(const float4*)&sc[r][c];
      float4 mv = *(const float4*)(mrow + c);
      sv.x += gamma * mv.x; sv.y += gamma * mv.y;
      sv.z += gamma * mv.z; sv.w += gamma * mv.w;
      *(float4*)&sc[r][c] = sv;
      mx = fmaxf(mx, fmaxf(fmaxf(sv.x, sv.y), fmaxf(sv.z, sv.w)));
    }
#pragma unroll
    for (int off = 16; off > 0; off >>= 1) mx = fmaxf(mx, __shfl_xor(mx, off, 32));
    float sum = 0.f;
    for (int c = lane * 4; c < S; c += 128) {
      float4 sv = *(const float4*)&sc[r][c];
      sv.x = __expf(sv.x - mx); sv.y = __expf(sv.y - mx);
      sv.z = __expf(sv.z - mx); sv.w = __expf(sv.w - mx);
      *(float4*)&sc[r][c] = sv;
      sum += (sv.x + sv.y) + (sv.z + sv.w);
    }
#pragma unroll
    for (int off = 16; off > 0; off >>= 1) sum += __shfl_xor(sum, off, 32);
    if (lane == 0) rowinv[r] = 1.0f / sum;
  }
  __syncthreads();

  // ---- 4) write attention matrix (float4, coalesced) ----
  {
    float* ab = attn_out + (bh * S + q0) * S;
    for (int i = threadIdx.x; i < 16 * (S / 4); i += 256) {
      int r = i >> 9, c4 = (i & 511) << 2;
      float4 p = *(const float4*)&sc[r][c4];
      float sinv = rowinv[r];
      p.x *= sinv; p.y *= sinv; p.z *= sinv; p.w *= sinv;
      *(float4*)(ab + (size_t)r * S + c4) = p;
    }
  }

  // ---- 5) out[16][64] = attn @ V : wave = (khalf, ntile); split-K reduce ----
  const int nt = wave & 3;
  const int kh = wave >> 2;
  const __bf16* vrow = Vt + ((bh << 6) + nt * 16 + m) * S;
  v8f oacc = {};
#pragma unroll 4
  for (int ks = kh * 32; ks < (kh + 1) * 32; ++ks) {
    const int k0 = ks * 32;
    const float* srow = &sc[m][k0];
    float4 p0 = *(const float4*)(srow + 16 * g);
    float4 p1 = *(const float4*)(srow + 16 * g + 4);
    float4 p2 = *(const float4*)(srow + 16 * g + 8);
    float4 p3 = *(const float4*)(srow + 16 * g + 12);
    v16bf ap = cvt16(p0, p1, p2, p3);
    v16bf bv = *(const v16bf*)(vrow + k0 + 16 * g);
    oacc = wmma_bf16(ap, bv, oacc);
  }
  __syncthreads();
  if (kh == 1) {
#pragma unroll
    for (int i = 0; i < 8; ++i) outbuf[mb + i][nt * 16 + m] = oacc[i];
  }
  __syncthreads();
  if (kh == 0) {
#pragma unroll
    for (int i = 0; i < 8; ++i) {
      float v = (oacc[i] + outbuf[mb + i][nt * 16 + m]) * rowinv[mb + i];
      out[(size_t)(bb * S + q0 + mb + i) * (H * DV) + h * DV + nt * 16 + m] = v;
    }
  }
}

// ---------------------------------------------------------------------------
extern "C" void kernel_launch(void* const* d_in, const int* in_sizes, int n_in,
                              void* d_out, int out_size, void* d_ws, size_t ws_size,
                              hipStream_t stream) {
  (void)in_sizes; (void)n_in; (void)out_size; (void)ws_size;
  const float* q    = (const float*)d_in[0];
  const float* k    = (const float*)d_in[1];
  const float* v    = (const float*)d_in[2];
  const float* mask = (const float*)d_in[3];
  const float* Wq   = (const float*)d_in[4];
  const float* Wk   = (const float*)d_in[5];
  const float* Wv   = (const float*)d_in[6];
  const float* gm   = (const float*)d_in[7];

  const size_t headElems = (size_t)B * H * S * DK;
  const size_t wElems    = (size_t)D * HD;
  __bf16* Qh  = (__bf16*)d_ws;
  __bf16* Kh  = Qh + headElems;
  __bf16* Vt  = Kh + headElems;                      // transposed [b,h,d,S]
  __bf16* Wqt = Vt + headElems;
  __bf16* Wkt = Wqt + wElems;
  __bf16* Wvt = Wkt + wElems;

  float* out  = (float*)d_out;                        // [B,S,H*DV]
  float* attn = out + (size_t)B * S * H * DV;         // [B,H,S,S]

  dim3 tgrid(D / 64, HD / 64);
  transpose_w<<<tgrid, 256, 0, stream>>>(Wq, Wqt);
  transpose_w<<<tgrid, 256, 0, stream>>>(Wk, Wkt);
  transpose_w<<<tgrid, 256, 0, stream>>>(Wv, Wvt);

  dim3 pgrid(B * S / 16, HD / 64);
  proj_kernel<<<pgrid, 128, 0, stream>>>(q, Wqt, Qh, 0.125f, 0);
  proj_kernel<<<pgrid, 128, 0, stream>>>(k, Wkt, Kh, 1.0f, 0);
  proj_kernel<<<pgrid, 128, 0, stream>>>(v, Wvt, Vt, 1.0f, 1);

  dim3 agrid(S / 16, H, B);
  attn_kernel<<<agrid, 256, 0, stream>>>(Qh, Kh, Vt, mask, gm, out, attn);
}
